// BertWithWordLee_79121887526919
// MI455X (gfx1250) — matrix operand order repaired and checked by reading, not compiled, not run on hardware
//
#include <hip/hip_runtime.h>

typedef __bf16 bf16_t;
typedef __attribute__((ext_vector_type(16))) __bf16 v16bf;
typedef __attribute__((ext_vector_type(8)))  __bf16 v8bf;
typedef __attribute__((ext_vector_type(4)))  __bf16 v4bf;
typedef __attribute__((ext_vector_type(8)))  float  v8f;
typedef __attribute__((ext_vector_type(4)))  float  v4f;

#define B_     16
#define N_     512
#define L_     2048
#define DIM_   768
#define MW_    8
#define CE_    64
#define VOCAB_ 1024
#define H_     128

// ---------------------------------------------------------------------------
// WMMA fragment loaders (CDNA5 16x16x32 bf16 layouts, wave32)
// A (16xK=32, 16-bit): lane l<16 -> row M=l, K = {0..7} u {16..23};
//                      lane l>=16 -> row M=l-16, K = {8..15} u {24..31}
// B (K=32x16, 16-bit): lane l<16 -> col N=l, K = {0..7} u {16..23};
//                      lane l>=16 -> col N=l-16, K = {8..15} u {24..31}
// ---------------------------------------------------------------------------
__device__ __forceinline__ v16bf load_a_frag(const bf16_t* base, int row_stride, int lane) {
    int r  = lane & 15;
    int kA = (lane < 16) ? 0 : 8;
    const bf16_t* p = base + r * row_stride;
    v8bf lo = *(const v8bf*)(p + kA);
    v8bf hi = *(const v8bf*)(p + kA + 16);
    v16bf a;
#pragma unroll
    for (int i = 0; i < 8; ++i) { a[i] = lo[i]; a[8 + i] = hi[i]; }
    return a;
}

__device__ __forceinline__ v16bf load_b_frag(const bf16_t* base, int row_stride, int lane) {
    int n = lane & 15;
    int k = (lane < 16) ? 0 : 8;
    const bf16_t* p = base + n * row_stride + k;
    v8bf lo = *(const v8bf*)(p);
    v8bf hi = *(const v8bf*)(p + 16);
    v16bf b;
#pragma unroll
    for (int i = 0; i < 8; ++i) { b[i] = lo[i]; b[8 + i] = hi[i]; }
    return b;
}

__device__ __forceinline__ float fast_tanh(float x) {
#if __has_builtin(__builtin_amdgcn_tanhf)
    return __builtin_amdgcn_tanhf(x);
#elif __has_builtin(__builtin_amdgcn_tanh_f32)
    return __builtin_amdgcn_tanh_f32(x);
#else
    return tanhf(x);
#endif
}

__device__ __forceinline__ float fast_rcp(float x) {
#if __has_builtin(__builtin_amdgcn_rcpf)
    return __builtin_amdgcn_rcpf(x);
#else
    return 1.0f / x;
#endif
}

__device__ __forceinline__ float sigmoidf_(float v) {
    return fast_rcp(1.0f + __expf(-v));
}

__device__ __forceinline__ v8f wmma_bf16(v16bf a, v16bf b, v8f c) {
    return __builtin_amdgcn_wmma_f32_16x16x32_bf16(
        false, a, false, b, (short)0, c, false, false);
}

// ---------------------------------------------------------------------------
// fp32 -> bf16 conversion
// ---------------------------------------------------------------------------
__global__ __launch_bounds__(256) void cvt_f32_bf16(const float* __restrict__ src,
                                                    bf16_t* __restrict__ dst, int n) {
    int i = blockIdx.x * 256 + threadIdx.x;
    if (i < n) dst[i] = (bf16_t)src[i];
}

// ---------------------------------------------------------------------------
// Generic bf16 WMMA GEMM: C[M,N] = A[M,K] @ B[K,N] + bias[N]   (fp32 out)
// block = 256 threads (8 waves); tile 128(M) x 64(N), K chunk = 32.
// Double-buffered staging: A via async global->LDS (ASYNCcnt), B via regs.
// Fragment loads batched ahead of WMMAs; epilogue staged via LDS, b128 stores.
// ---------------------------------------------------------------------------
__global__ __launch_bounds__(256) void gemm_bf16_wmma(
    const bf16_t* __restrict__ A, const bf16_t* __restrict__ Bm,
    const float* __restrict__ bias, float* __restrict__ C,
    int Mdim, int Ndim, int Kdim)
{
    __shared__ __align__(16) char smem[128 * 68 * 4];   // 34,816 B
    bf16_t* As = (bf16_t*)smem;                  // 2 x (128*32) bf16 = 16 KB
    bf16_t* Bs = (bf16_t*)(smem + 16384);        // 2 x (64*32)  bf16 = 8 KB
    float*  Cs = (float*)smem;                   // epilogue reuse: 128 x 68 f32

    const int tid  = threadIdx.x;
    const int wave = tid >> 5;
    const int lane = tid & 31;
    const int gRow0 = blockIdx.y * 128;
    const int gCol0 = blockIdx.x * 64;

    v8f acc[4];
#pragma unroll
    for (int nt = 0; nt < 4; ++nt) {
        float bv = bias[gCol0 + nt * 16 + (lane & 15)];
#pragma unroll
        for (int r = 0; r < 8; ++r) acc[nt][r] = bv;
    }

    // ---- prologue: stage K-chunk 0 into buffer 0 ----
#pragma unroll
    for (int j = 0; j < 2; ++j) {
        int off = (tid + j * 256) * 8;           // 512 chunks of 8 bf16
        int r = off >> 5, c = off & 31;
        unsigned lds = (unsigned)(uintptr_t)(As + r * 32 + c);
        const bf16_t* g = A + (size_t)(gRow0 + r) * Kdim + c;
        asm volatile("global_load_async_to_lds_b128 %0, %1, off"
                     :: "v"(lds), "v"(g) : "memory");
    }
#pragma unroll
    for (int j = 0; j < 8; ++j) {
        int i  = tid + j * 256;
        int kk = i >> 6, n = i & 63;
        Bs[n * 32 + kk] = Bm[(size_t)kk * Ndim + gCol0 + n];
    }
    asm volatile("s_wait_asynccnt 0" ::: "memory");
    __syncthreads();

    const int nIter = Kdim >> 5;
    for (int it = 0; it < nIter; ++it) {
        bf16_t* Acur = As + (it & 1) * (128 * 32);
        bf16_t* Bcur = Bs + (it & 1) * (64 * 32);
        bf16_t* Anxt = As + ((it + 1) & 1) * (128 * 32);
        bf16_t* Bnxt = Bs + ((it + 1) & 1) * (64 * 32);
        const int kn = (it + 1) << 5;
        const bool have_next = (kn < Kdim);

        // prefetch next K-chunk: A async into LDS, B into registers
        bf16_t breg[8];
        if (have_next) {
#pragma unroll
            for (int j = 0; j < 2; ++j) {
                int off = (tid + j * 256) * 8;
                int r = off >> 5, c = off & 31;
                unsigned lds = (unsigned)(uintptr_t)(Anxt + r * 32 + c);
                const bf16_t* g = A + (size_t)(gRow0 + r) * Kdim + kn + c;
                asm volatile("global_load_async_to_lds_b128 %0, %1, off"
                             :: "v"(lds), "v"(g) : "memory");
            }
#pragma unroll
            for (int j = 0; j < 8; ++j) {
                int i  = tid + j * 256;
                int kk = i >> 6, n = i & 63;
                breg[j] = Bm[(size_t)(kn + kk) * Ndim + gCol0 + n];
            }
        }

        // compute on current buffers: batch all fragment loads, then WMMAs
        v16bf a = load_a_frag(Acur + (wave * 16) * 32, 32, lane);
        v16bf bfr[4];
#pragma unroll
        for (int nt = 0; nt < 4; ++nt)
            bfr[nt] = load_b_frag(Bcur + (nt * 16) * 32, 32, lane);
#pragma unroll
        for (int nt = 0; nt < 4; ++nt)
            acc[nt] = wmma_bf16(a, bfr[nt], acc[nt]);

        // commit B registers to next buffer (transposed [n][k])
        if (have_next) {
#pragma unroll
            for (int j = 0; j < 8; ++j) {
                int i  = tid + j * 256;
                int kk = i >> 6, n = i & 63;
                Bnxt[n * 32 + kk] = breg[j];
            }
        }
        asm volatile("s_wait_asynccnt 0" ::: "memory");
        __syncthreads();
    }

    // ---- epilogue: stage tile in LDS (stride 68 keeps 16B alignment) ----
#pragma unroll
    for (int nt = 0; nt < 4; ++nt) {
        int col   = nt * 16 + (lane & 15);
        int rbase = wave * 16 + ((lane < 16) ? 0 : 8);
#pragma unroll
        for (int r = 0; r < 8; ++r)
            Cs[(rbase + r) * 68 + col] = acc[nt][r];
    }
    __syncthreads();
#pragma unroll
    for (int j = 0; j < 8; ++j) {
        int idx = tid + j * 256;                  // 2048 float4 chunks
        int r = idx >> 4, c4 = (idx & 15) << 2;
        *(v4f*)&C[(size_t)(gRow0 + r) * Ndim + gCol0 + c4] =
            *(const v4f*)&Cs[r * 68 + c4];
    }
}

// ---------------------------------------------------------------------------
// Char embedding gather into x[:, :, 0:64] (bf16) and zero x[:, :, 64:128]
// ---------------------------------------------------------------------------
__global__ __launch_bounds__(128) void embed_zero_kernel(
    const int* __restrict__ ids, const float* __restrict__ emb,
    bf16_t* __restrict__ x)
{
    int row = blockIdx.x;        // b*L + l
    int c   = threadIdx.x;       // 0..127
    bf16_t v = (bf16_t)0.f;
    if (c < CE_) {
        int id = ids[row];
        v = (bf16_t)emb[id * CE_ + c];
    }
    x[(size_t)row * 128 + c] = v;
}

// ---------------------------------------------------------------------------
// Ragged compaction: rep[b,n,m,:] -> x[b, dest, 64:128] (bf16)
// ---------------------------------------------------------------------------
__global__ __launch_bounds__(256) void gather_pad_kernel(
    const float* __restrict__ rep, const int* __restrict__ lens,
    bf16_t* __restrict__ x)
{
    __shared__ int offs[N_];
    int b = blockIdx.x;
    const int* wl = lens + b * N_;

    if (threadIdx.x == 0) {
        int s = 0;
        for (int n = 0; n < N_; ++n) {
            offs[n] = s;
            int l = wl[n]; if (l > MW_) l = MW_;
            s += l;
        }
    }
    __syncthreads();

    for (int n = threadIdx.x; n < N_; n += 256) {
        int len = wl[n]; if (len > MW_) len = MW_;
        int off = offs[n];
        const float* src = rep + (size_t)(b * N_ + n) * (MW_ * CE_);
        for (int m = 0; m < len; ++m) {
            int dest = off + m;
            if (dest >= L_) break;
            bf16_t* dst = x + (size_t)(b * L_ + dest) * 128 + CE_;
            const float* s = src + m * CE_;
            for (int c = 0; c < CE_; ++c) dst[c] = (bf16_t)s[c];
        }
    }
}

// ---------------------------------------------------------------------------
// Persistent LSTM: 512 threads = 16 waves, ALL weight fragments
// register-resident (128 VGPRs/lane). Wave w<8 owns H-tile w, gates (i, g~);
// wave w+8 owns same H-tile, gates (f, o) + cell state. Gate exchange via an
// 8 KB f32 LDS buffer; 2 workgroup barriers per step; LDS weight traffic = 0.
// ---------------------------------------------------------------------------
__global__ __launch_bounds__(512) void lstm_kernel(
    const bf16_t* __restrict__ x,      // [B, L, 128] bf16
    const bf16_t* __restrict__ Wih,    // [4H=512, 128] bf16
    const bf16_t* __restrict__ Whh,    // [512, 128] bf16
    const float* __restrict__ b_ih,    // [512]
    const float* __restrict__ b_hh,    // [512]
    bf16_t* __restrict__ hs)           // [B, L, 128] bf16
{
    __shared__ __align__(16) bf16_t sH[16 * 128];   // h_t, 4 KB
    __shared__ __align__(16) float  sM[16 * 128];   // sigmoid(i)*tanh(g~), 8 KB

    const int tid  = threadIdx.x;
    const int wv   = tid >> 5;          // 0..15
    const int lane = tid & 31;
    const int grp  = wv >> 3;           // 0: (i, g~)   1: (f, o)
    const int jt   = wv & 7;            // H-tile index
    const int g0row = (grp == 0) ? 0 : 128;     // i  or f
    const int g1row = (grp == 0) ? 256 : 384;   // g~ or o

    // ---- load weight fragments into registers (one-time) ----
    v16bf wx0[4], wx1[4], wh0[4], wh1[4];
#pragma unroll
    for (int k = 0; k < 4; ++k) {
        wx0[k] = load_b_frag(Wih + (size_t)(g0row + jt * 16) * 128 + k * 32, 128, lane);
        wx1[k] = load_b_frag(Wih + (size_t)(g1row + jt * 16) * 128 + k * 32, 128, lane);
        wh0[k] = load_b_frag(Whh + (size_t)(g0row + jt * 16) * 128 + k * 32, 128, lane);
        wh1[k] = load_b_frag(Whh + (size_t)(g1row + jt * 16) * 128 + k * 32, 128, lane);
    }
    const int ncol = jt * 16 + (lane & 15);
    const float bias0 = b_ih[g0row + ncol] + b_hh[g0row + ncol];
    const float bias1 = b_ih[g1row + ncol] + b_hh[g1row + ncol];

    for (int i = tid; i < 16 * 128; i += 512) sH[i] = (bf16_t)0.f;
    __syncthreads();

    v8f c_acc = {};   // cell state (grp==1 waves only)

    for (int t = 0; t < L_; ++t) {
        // issue x_t fragment loads first (hidden under the h-GEMM)
        v16bf ax[4];
#pragma unroll
        for (int k = 0; k < 4; ++k)
            ax[k] = load_a_frag(x + (size_t)t * 128 + k * 32, L_ * 128, lane);
        if (t + 8 < L_)
            __builtin_prefetch(x + (size_t)(t + 8) * 128, 0, 0);

        v8f g0, g1;
#pragma unroll
        for (int r = 0; r < 8; ++r) { g0[r] = bias0; g1[r] = bias1; }

        // recurrent GEMM (sH valid from previous barrier); batch frag loads
        v16bf ah[4];
#pragma unroll
        for (int k = 0; k < 4; ++k)
            ah[k] = load_a_frag(&sH[k * 32], 128, lane);
#pragma unroll
        for (int k = 0; k < 4; ++k) {
            g0 = wmma_bf16(ah[k], wh0[k], g0);
            g1 = wmma_bf16(ah[k], wh1[k], g1);
        }
        // input GEMM (ax arrived by now)
#pragma unroll
        for (int k = 0; k < 4; ++k) {
            g0 = wmma_bf16(ax[k], wx0[k], g0);
            g1 = wmma_bf16(ax[k], wx1[k], g1);
        }

        const int rbase = (lane < 16) ? 0 : 8;
        if (grp == 0) {
            // m = sigmoid(i) * tanh(g~) -> exchange buffer
#pragma unroll
            for (int r = 0; r < 8; ++r)
                sM[(rbase + r) * 128 + ncol] = sigmoidf_(g0[r]) * fast_tanh(g1[r]);
        }
        __syncthreads();   // (1) m visible; all sH reads of step t done
        if (grp == 1) {
#pragma unroll
            for (int r = 0; r < 8; ++r) {
                float m  = sM[(rbase + r) * 128 + ncol];
                float cc = sigmoidf_(g0[r]) * c_acc[r] + m;   // f-gate * c + i*g~
                c_acc[r] = cc;
                float hv = sigmoidf_(g1[r]) * fast_tanh(cc);  // o-gate * tanh(c)
                sH[(rbase + r) * 128 + ncol] = (bf16_t)hv;
            }
        }
        __syncthreads();   // (2) new h stable in LDS

        // cooperative coalesced copy sH -> hs[:, t, :]  (8B per thread)
        {
            int idx = tid * 4;                 // 2048 bf16 total
            int row = idx >> 7, col = idx & 127;
            *(v4bf*)&hs[((size_t)row * L_ + t) * 128 + col] =
                *(const v4bf*)&sH[idx];
        }
    }
}

// ---------------------------------------------------------------------------
extern "C" void kernel_launch(void* const* d_in, const int* in_sizes, int n_in,
                              void* d_out, int out_size, void* d_ws, size_t ws_size,
                              hipStream_t stream) {
    const float* dec_hidden = (const float*)d_in[0];
    const int*   ids        = (const int*)d_in[1];
    const int*   lens       = (const int*)d_in[2];
    const float* char_emb   = (const float*)d_in[3];
    const float* W_proj     = (const float*)d_in[4];
    const float* b_proj     = (const float*)d_in[5];
    const float* W_ih       = (const float*)d_in[6];
    const float* W_hh       = (const float*)d_in[7];
    const float* b_ih       = (const float*)d_in[8];
    const float* b_hh       = (const float*)d_in[9];
    const float* W_pred     = (const float*)d_in[10];
    const float* b_pred     = (const float*)d_in[11];
    float* out = (float*)d_out;

    char*  ws  = (char*)d_ws;
    size_t off = 0;
    auto alloc = [&](size_t bytes) -> void* {
        void* p = ws + off;
        off += (bytes + 255) & ~(size_t)255;
        return p;
    };
    const size_t nDec  = (size_t)B_ * N_ * DIM_;          // 6,291,456
    const size_t nWp   = (size_t)DIM_ * (CE_ * MW_);      // 393,216
    const size_t nWpr  = (size_t)H_ * VOCAB_;             // 131,072
    const size_t nWg   = (size_t)(4 * H_) * (2 * CE_);    // 65,536 (W_ih/W_hh)
    bf16_t* A_dec   = (bf16_t*)alloc(nDec * 2);
    bf16_t* Wp_b    = (bf16_t*)alloc(nWp * 2);
    bf16_t* Wpred_b = (bf16_t*)alloc(nWpr * 2);
    bf16_t* Wih_b   = (bf16_t*)alloc(nWg * 2);
    bf16_t* Whh_b   = (bf16_t*)alloc(nWg * 2);
    float*  rep     = (float*) alloc((size_t)B_ * N_ * MW_ * CE_ * 4);  // [8192,512]
    bf16_t* x       = (bf16_t*)alloc((size_t)B_ * L_ * 128 * 2);
    bf16_t* hs      = (bf16_t*)alloc((size_t)B_ * L_ * 128 * 2);

    // 1) one-time bf16 conversions
    cvt_f32_bf16<<<(int)((nDec + 255) / 256), 256, 0, stream>>>(dec_hidden, A_dec, (int)nDec);
    cvt_f32_bf16<<<(int)((nWp  + 255) / 256), 256, 0, stream>>>(W_proj, Wp_b, (int)nWp);
    cvt_f32_bf16<<<(int)((nWpr + 255) / 256), 256, 0, stream>>>(W_pred, Wpred_b, (int)nWpr);
    cvt_f32_bf16<<<(int)((nWg  + 255) / 256), 256, 0, stream>>>(W_ih, Wih_b, (int)nWg);
    cvt_f32_bf16<<<(int)((nWg  + 255) / 256), 256, 0, stream>>>(W_hh, Whh_b, (int)nWg);

    // 2) projection GEMM: rep[8192,512] = dec @ W_proj + b_proj
    {
        dim3 grid((CE_ * MW_) / 64, (B_ * N_) / 128);     // (8, 64)
        gemm_bf16_wmma<<<grid, 256, 0, stream>>>(A_dec, Wp_b, b_proj, rep,
                                                 B_ * N_, CE_ * MW_, DIM_);
    }

    // 3) char-embed gather into x[:,:,0:64], zero x[:,:,64:128]
    embed_zero_kernel<<<B_ * L_, 128, 0, stream>>>(ids, char_emb, x);

    // 4) ragged compaction into x[:,:,64:128]
    gather_pad_kernel<<<B_, 256, 0, stream>>>(rep, lens, x);

    // 5) persistent LSTM: 16 waves, register-resident weights
    lstm_kernel<<<1, 512, 0, stream>>>(x, Wih_b, Whh_b, b_ih, b_hh, hs);

    // 6) output projection: out[32768,1024] = hs @ W_pred + b_pred
    {
        dim3 grid(VOCAB_ / 64, (B_ * L_) / 128);          // (16, 256)
        gemm_bf16_wmma<<<grid, 256, 0, stream>>>(hs, Wpred_b, b_pred, out,
                                                 B_ * L_, VOCAB_, H_);
    }
}
